// BTTLayer_19138374271635
// MI455X (gfx1250) — matrix operand order repaired and checked by reading, not compiled, not run on hardware
//
#include <hip/hip_runtime.h>
#include <hip/hip_bf16.h>

// ---------------------------------------------------------------------------
// BTT layer for MI455X (gfx1250, wave32, WMMA).
//   Stage 1: t[b,m2,n1,r]  = sum_m1 x[b,m2,m1] * W1[m2,m1,n1*4+r]
//   Stage 2: y[b,n1,n2]    = sum_{m2,r} t[b,n1,m2*4+r] * W2[n1,m2*4+r,n2]
// f32 math emulated as 3x bf16 WMMA (hi/lo split). Splits/transposes hoisted
// out of the K loop. Square-ish wave tiles balance A/B fragment reuse:
// stage 1 = 32x64/wave (1.0 ds_load per wmma), stage 2 = 32x32/wave (1.33).
// ---------------------------------------------------------------------------

typedef __attribute__((ext_vector_type(16))) __bf16 v16bf;
typedef __attribute__((ext_vector_type(8)))  __bf16 v8bf;
typedef __attribute__((ext_vector_type(8)))  float  v8f;

#define B_TOT  4096
#define D_IN_  4096
#define D_OUT_ 4096

// f32 pair -> packed bf16-hi dword + packed bf16-lo dword (truncation split;
// lo captures the exact remainder, so hi*hi + hi*lo + lo*hi ~ f32 product).
__device__ __forceinline__ void split_pack(float f0, float f1,
                                           unsigned& hp, unsigned& lp) {
  unsigned u0 = __builtin_bit_cast(unsigned, f0);
  unsigned u1 = __builtin_bit_cast(unsigned, f1);
  unsigned h0 = u0 & 0xFFFF0000u;
  unsigned h1 = u1 & 0xFFFF0000u;
  hp = (h0 >> 16) | h1;
  float l0 = f0 - __builtin_bit_cast(float, h0);
  float l1 = f1 - __builtin_bit_cast(float, h1);
  lp = (__builtin_bit_cast(unsigned, l0) >> 16) |
       (__builtin_bit_cast(unsigned, l1) & 0xFFFF0000u);
}

// A fragment (16x32 bf16, wave32): per lane, halves 0..7 = K kb..kb+7 and
// halves 8..15 = K kb+16..kb+23 -> two contiguous 16B LDS loads.
__device__ __forceinline__ v16bf load_a_frag(const __bf16* rowp, int kb) {
  v8bf a0 = *(const v8bf*)(rowp + kb);
  v8bf a1 = *(const v8bf*)(rowp + kb + 16);
  return __builtin_shufflevector(a0, a1, 0, 1, 2, 3, 4, 5, 6, 7,
                                 8, 9, 10, 11, 12, 13, 14, 15);
}

// B fragment (32x16 bf16, wave32): per lane (column), halves 0..7 = K
// kb..kb+7 and halves 8..15 = K kb+16..kb+23 (K-contiguous LDS layout).
__device__ __forceinline__ v16bf load_b_frag(const __bf16* colp, int kb) {
  v8bf b0 = *(const v8bf*)(colp + kb);
  v8bf b1 = *(const v8bf*)(colp + kb + 16);
  return __builtin_shufflevector(b0, b1, 0, 1, 2, 3, 4, 5, 6, 7,
                                 8, 9, 10, 11, 12, 13, 14, 15);
}

__device__ __forceinline__ v8f wmma3(v16bf ah, v16bf al, v16bf bh, v16bf bl, v8f c) {
  c = __builtin_amdgcn_wmma_f32_16x16x32_bf16(false, ah, false, bh, (short)0, c, false, false);
  c = __builtin_amdgcn_wmma_f32_16x16x32_bf16(false, ah, false, bl, (short)0, c, false, false);
  c = __builtin_amdgcn_wmma_f32_16x16x32_bf16(false, al, false, bh, (short)0, c, false, false);
  return c;
}

// ------------- Prep: split + transpose weights into bf16 hi/lo -------------
// w1T[m2][c][k] = split(W1[m2][k][c]);  w2T[n1][n2][d] = split(W2[n1][d][n2])
__global__ void __launch_bounds__(256)
btt_prep(const float* __restrict__ W1, const float* __restrict__ W2,
         unsigned* __restrict__ w1h, unsigned* __restrict__ w1l,
         unsigned* __restrict__ w2h, unsigned* __restrict__ w2l) {
  unsigned g = blockIdx.x * 256u + threadIdx.x;
  if (g < 524288u) {            // W1: m2(64) x c(256) x kp(32)
    unsigned m2 = g >> 13, rem = g & 8191u;
    unsigned c = rem >> 5, kp = rem & 31u;
    const float* p = W1 + (size_t)m2 * 16384 + (2 * kp) * 256 + c;
    unsigned hp, lp;
    split_pack(p[0], p[256], hp, lp);
    w1h[g] = hp;
    w1l[g] = lp;
  } else {                      // W2: n1(64) x n2(64) x dp(128)
    unsigned gg = g - 524288u;
    unsigned n1 = gg >> 13, rem = gg & 8191u;
    unsigned n2 = rem >> 7, dp = rem & 127u;
    const float* p = W2 + (size_t)n1 * 16384 + (2 * dp) * 64 + n2;
    unsigned hp, lp;
    split_pack(p[0], p[64], hp, lp);
    w2h[gg] = hp;
    w2l[gg] = lp;
  }
}

// ---------------- Stage 1: per-m2 GEMM (64b x 64k) @ (64k x 256c) ------------
// 8 waves tiled 2(b-half) x 4(c-quarter); wave tile 32x64 (8 accum tiles).
// Output tile written block-contiguous: t1[(m2*64+bt)][b_local][c], 64KB/block.
__global__ void __launch_bounds__(256)
btt_stage1(const float* __restrict__ x,
           const __bf16* __restrict__ w1h, const __bf16* __restrict__ w1l,
           float* __restrict__ t_ws) {
  extern __shared__ char smem[];
  __bf16* xh = (__bf16*)smem;        // 64 x 64  (b x k)      8 KB
  __bf16* xl = xh + 4096;            //                        8 KB
  __bf16* wh = xl + 4096;            // 256 x 64 (c x k)      32 KB
  __bf16* wl = wh + 16384;           //                       32 KB

  const int m2 = blockIdx.x;
  const int bt = blockIdx.y;
  const int tid = threadIdx.x;

  // x tile: split on the fly (each x element touched by exactly one block).
  for (int p = tid; p < 2048; p += 256) {          // 2048 float2 pairs
    int row = p >> 5, pr = p & 31;
    float2 v = *(const float2*)(x + (size_t)(bt * 64 + row) * D_IN_ +
                                m2 * 64 + pr * 2);
    unsigned hp, lp;
    split_pack(v.x, v.y, hp, lp);
    ((unsigned*)xh)[p] = hp;
    ((unsigned*)xl)[p] = lp;
  }
  // weights: contiguous bf16 copy (pre-split, pre-transposed by btt_prep)
  {
    const uint4* sh = (const uint4*)(w1h + (size_t)m2 * 16384);
    const uint4* sl = (const uint4*)(w1l + (size_t)m2 * 16384);
    for (int i = tid; i < 2048; i += 256) {
      ((uint4*)wh)[i] = sh[i];
      ((uint4*)wl)[i] = sl[i];
    }
  }
  __syncthreads();

  const int wave = tid >> 5, lane = tid & 31;
  const int wbh = wave >> 2;         // b half (32 rows)     0..1
  const int wcq = wave & 3;          // c quarter (64 cols)  0..3
  const int laneHi = lane >> 4;
  const int lrow = lane & 15;

  v8f zero = {0.f, 0.f, 0.f, 0.f, 0.f, 0.f, 0.f, 0.f};
  v8f acc[2][4];
#pragma unroll
  for (int rt = 0; rt < 2; ++rt)
#pragma unroll
    for (int ct = 0; ct < 4; ++ct) acc[rt][ct] = zero;

  const __bf16* arh[2];
  const __bf16* arl[2];
#pragma unroll
  for (int rt = 0; rt < 2; ++rt) {
    arh[rt] = xh + (wbh * 32 + rt * 16 + lrow) * 64;
    arl[rt] = xl + (wbh * 32 + rt * 16 + lrow) * 64;
  }

#pragma unroll
  for (int kk = 0; kk < 64; kk += 32) {
    const int kbA = kk + laneHi * 8;
    const int kbB = kk + laneHi * 8;
    v16bf ah[2], al[2];
#pragma unroll
    for (int rt = 0; rt < 2; ++rt) {
      ah[rt] = load_a_frag(arh[rt], kbA);
      al[rt] = load_a_frag(arl[rt], kbA);
    }
#pragma unroll
    for (int ct = 0; ct < 4; ++ct) {
      const int col = wcq * 64 + ct * 16 + lrow;
      v16bf bh = load_b_frag(wh + col * 64, kbB);
      v16bf bl = load_b_frag(wl + col * 64, kbB);
#pragma unroll
      for (int rt = 0; rt < 2; ++rt)
        acc[rt][ct] = wmma3(ah[rt], al[rt], bh, bl, acc[rt][ct]);
    }
  }

  // Block-contiguous store: one per-lane base, all else immediate offsets.
  // C/D layout: VGPR d -> M=d (lanes 0-15) / d+8 (lanes 16-31); N = lane&15.
  float* tp = t_ws + (size_t)(m2 * 64 + bt) * 16384 +
              (wbh * 32 + laneHi * 8) * 256 + wcq * 64 + lrow;
#pragma unroll
  for (int rt = 0; rt < 2; ++rt)
#pragma unroll
    for (int ct = 0; ct < 4; ++ct)
#pragma unroll
      for (int d = 0; d < 8; ++d)
        tp[rt * 4096 + d * 256 + ct * 16] = acc[rt][ct][d];
}

// ---------------- Stage 2: per-n1 GEMM (128b x 256k) @ (256k x 64n2) ---------
// Block tile 128b x 64n2 (192KB LDS); 8 waves tiled 4(b-quarter) x 2(n2-half);
// wave tile 32x32 (4 accum tiles). LDS fill gathers [m2] scatter as float4.
__global__ void __launch_bounds__(256)
btt_stage2(const float* __restrict__ t_ws,
           const __bf16* __restrict__ w2h, const __bf16* __restrict__ w2l,
           float* __restrict__ y) {
  extern __shared__ char smem[];
  __bf16* ah_s = (__bf16*)smem;      // 128 x 256 (b x d)  64 KB
  __bf16* al_s = ah_s + 32768;       //                    64 KB
  __bf16* bh_s = al_s + 32768;       // 64 x 256 (n2 x d)  32 KB
  __bf16* bl_s = bh_s + 16384;       //                    32 KB

  const int n1 = blockIdx.x;
  const int bt = blockIdx.y;         // 128-row tile, 0..31
  const int tid = threadIdx.x;

  // Gather A tile: as[b][m2*4+r] = t1[(m2*64 + 2bt + b/64)][b%64][n1*4+r].
  for (int p = tid; p < 8192; p += 256) {          // p = m2*128 + b
    int m2 = p >> 7, b = p & 127;
    float4 v = *(const float4*)(t_ws +
                (size_t)(m2 * 64 + bt * 2 + (b >> 6)) * 16384 +
                (b & 63) * 256 + n1 * 4);
    unsigned h0, l0, h1, l1;
    split_pack(v.x, v.y, h0, l0);
    split_pack(v.z, v.w, h1, l1);
    int di = b * 128 + m2 * 2;                     // u32 index (row len 128)
    ((unsigned*)ah_s)[di]     = h0;
    ((unsigned*)ah_s)[di + 1] = h1;
    ((unsigned*)al_s)[di]     = l0;
    ((unsigned*)al_s)[di + 1] = l1;
  }
  {
    const uint4* sh = (const uint4*)(w2h + (size_t)n1 * 16384);
    const uint4* sl = (const uint4*)(w2l + (size_t)n1 * 16384);
    for (int i = tid; i < 2048; i += 256) {
      ((uint4*)bh_s)[i] = sh[i];
      ((uint4*)bl_s)[i] = sl[i];
    }
  }
  __syncthreads();

  const int wave = tid >> 5, lane = tid & 31;
  const int wq = wave >> 1;          // b quarter (32 rows of 128)  0..3
  const int wh2 = wave & 1;          // n2 half (32 cols)           0..1
  const int laneHi = lane >> 4;
  const int lrow = lane & 15;

  v8f zero = {0.f, 0.f, 0.f, 0.f, 0.f, 0.f, 0.f, 0.f};
  v8f acc[2][2];
#pragma unroll
  for (int rt = 0; rt < 2; ++rt)
#pragma unroll
    for (int ct = 0; ct < 2; ++ct) acc[rt][ct] = zero;

  const __bf16* arh[2];
  const __bf16* arl[2];
#pragma unroll
  for (int rt = 0; rt < 2; ++rt) {
    arh[rt] = ah_s + (wq * 32 + rt * 16 + lrow) * 256;
    arl[rt] = al_s + (wq * 32 + rt * 16 + lrow) * 256;
  }

#pragma unroll
  for (int kk = 0; kk < 256; kk += 32) {
    const int kbA = kk + laneHi * 8;
    const int kbB = kk + laneHi * 8;
    v16bf ah[2], al[2];
#pragma unroll
    for (int rt = 0; rt < 2; ++rt) {
      ah[rt] = load_a_frag(arh[rt], kbA);
      al[rt] = load_a_frag(arl[rt], kbA);
    }
#pragma unroll
    for (int ct = 0; ct < 2; ++ct) {
      const int col = wh2 * 32 + ct * 16 + lrow;
      v16bf bh = load_b_frag(bh_s + col * 256, kbB);
      v16bf bl = load_b_frag(bl_s + col * 256, kbB);
#pragma unroll
      for (int rt = 0; rt < 2; ++rt)
        acc[rt][ct] = wmma3(ah[rt], al[rt], bh, bl, acc[rt][ct]);
    }
  }

  // One per-lane base; rt/d/ct offsets fold into 24-bit immediates.
  float* yp = y + (size_t)(bt * 128 + wq * 32 + laneHi * 8) * D_OUT_ +
              n1 * 64 + wh2 * 32 + lrow;
#pragma unroll
  for (int rt = 0; rt < 2; ++rt)
#pragma unroll
    for (int ct = 0; ct < 2; ++ct)
#pragma unroll
      for (int d = 0; d < 8; ++d)
        yp[(rt * 16 + d) * D_OUT_ + ct * 16] = acc[rt][ct][d];
}

// ---------------------------------------------------------------------------
extern "C" void kernel_launch(void* const* d_in, const int* in_sizes, int n_in,
                              void* d_out, int out_size, void* d_ws, size_t ws_size,
                              hipStream_t stream) {
  const float* x  = (const float*)d_in[0];
  const float* W1 = (const float*)d_in[1];
  const float* W2 = (const float*)d_in[2];
  float* yv = (float*)d_out;

  // Workspace: [ t1 (f32, 256 MB) | w1h | w1l | w2h | w2l (bf16, 2 MB each) ]
  char* ws = (char*)d_ws;
  const size_t TBYTES = (size_t)4096 * 16384 * sizeof(float);
  const size_t WBYTES = (size_t)1048576 * sizeof(__bf16);
  float*  t_ws = (float*)ws;
  __bf16* w1h = (__bf16*)(ws + TBYTES);
  __bf16* w1l = (__bf16*)(ws + TBYTES + WBYTES);
  __bf16* w2h = (__bf16*)(ws + TBYTES + 2 * WBYTES);
  __bf16* w2l = (__bf16*)(ws + TBYTES + 3 * WBYTES);

  const int lds1 = (4096 + 4096 + 16384 + 16384) * 2;                // 80 KB
  const int lds2 = (32768 + 32768 + 16384 + 16384) * 2;              // 192 KB
  (void)hipFuncSetAttribute(reinterpret_cast<const void*>(btt_stage1),
                            hipFuncAttributeMaxDynamicSharedMemorySize, lds1);
  (void)hipFuncSetAttribute(reinterpret_cast<const void*>(btt_stage2),
                            hipFuncAttributeMaxDynamicSharedMemorySize, lds2);

  btt_prep<<<4096, 256, 0, stream>>>(W1, W2, (unsigned*)w1h, (unsigned*)w1l,
                                     (unsigned*)w2h, (unsigned*)w2l);
  dim3 grid1(64, B_TOT / 64);
  btt_stage1<<<grid1, 256, lds1, stream>>>(x, w1h, w1l, t_ws);
  dim3 grid2(64, B_TOT / 128);
  btt_stage2<<<grid2, 256, lds2, stream>>>(t_ws, w2h, w2l, yv);
}
